// ONLSTM_30915174597131
// MI455X (gfx1250) — matrix operand order repaired and checked by reading, not compiled, not run on hardware
//
#include <hip/hip_runtime.h>
#include <hip/hip_bf16.h>

typedef __bf16 bf16;
typedef __attribute__((ext_vector_type(16))) __bf16 v16bf;
typedef __attribute__((ext_vector_type(8)))  __bf16 v8bf;
typedef __attribute__((ext_vector_type(8)))  float  v8f;
typedef __attribute__((ext_vector_type(4)))  unsigned v4u;
typedef __attribute__((ext_vector_type(8)))  unsigned v8u;

union BF16x16 { v16bf v; v8bf h[2]; bf16 e[16]; };

#define B_SZ 64
#define T_SZ 1024
#define D_SZ 512
#define U_SZ 512
#define W_SZ 2080
#define NWG  64
#define GRP  16          // workgroups per barrier group (same batch quarter)

// workspace layout (bytes)
#define OFF_BAR 0ull                    // 4 groups * 1024 steps * 4B = 16384
#define OFF_H0  16384ull                // 64*512*2 = 65536
#define OFF_H1  (OFF_H0 + 65536ull)
#define OFF_RT  (OFF_H1 + 65536ull)     // 2080*512*2 = 2129920
#define OFF_XK  (OFF_RT + 2129920ull)   // 64*1024*2080*4 fp32

// ---------------------------------------------------------------------------
// recurrent_kernel (512 x 2080 fp32, row-major) -> bf16 transposed [w][k]
// ---------------------------------------------------------------------------
__global__ void onlstm_init_rt(const float* __restrict__ R, bf16* __restrict__ RT) {
  int idx = blockIdx.x * 256 + threadIdx.x;   // 2080*512, exact
  int w = idx >> 9;
  int k = idx & 511;
  RT[idx] = (bf16)R[(size_t)k * W_SZ + w];
}

// ---------------------------------------------------------------------------
// Phase 1: xk = x @ kernel + bias.  M=65536, N=2080, K=512.
// Block 256 thr (8 waves), tile 128M x 32N; wave tile 16M x 32N.
// ---------------------------------------------------------------------------
__launch_bounds__(256)
__global__ void onlstm_xk_gemm(const float* __restrict__ x, const float* __restrict__ Wk,
                               const float* __restrict__ bias, float* __restrict__ xk) {
  __shared__ bf16 Blds[32 * 40];          // [n][k], padded stride 40

  const int np    = blockIdx.x % 65;
  const int mt    = blockIdx.x / 65;
  const int n0    = np * 32;
  const int wid   = threadIdx.x >> 5;
  const int lane  = threadIdx.x & 31;
  const int m0    = mt * 128 + wid * 16;
  const int mrow  = m0 + (lane & 15);
  const int klaneA = (lane >> 4) * 8;     // A frag K chunk base
  const int klaneB = (lane >> 4) * 8;     // B frag K chunk base (same as A)
  const int nl_st = threadIdx.x & 31;
  const int kl_st = (threadIdx.x >> 5) * 4;

  v8f acc0 = {}; v8f acc1 = {};

  for (int kk = 0; kk < 16; ++kk) {
    #pragma unroll
    for (int i = 0; i < 4; ++i) {
      int kl = kl_st + i;
      Blds[nl_st * 40 + kl] = (bf16)Wk[(size_t)(kk * 32 + kl) * W_SZ + n0 + nl_st];
    }
    __syncthreads();

    // A fragment: K = klaneA+[0,8) and klaneA+16+[0,8)
    const float* px = x + (size_t)mrow * D_SZ + kk * 32 + klaneA;
    float4 a0 = *(const float4*)(px);
    float4 a1 = *(const float4*)(px + 4);
    float4 a2 = *(const float4*)(px + 16);
    float4 a3 = *(const float4*)(px + 20);
    float fa[16] = {a0.x,a0.y,a0.z,a0.w, a1.x,a1.y,a1.z,a1.w,
                    a2.x,a2.y,a2.z,a2.w, a3.x,a3.y,a3.z,a3.w};
    BF16x16 a;
    #pragma unroll
    for (int j = 0; j < 16; ++j) a.e[j] = (bf16)fa[j];

    // B fragments: lane holds column n, K = klaneB+[0,8) and klaneB+16+[0,8)
    BF16x16 b0, b1;
    const bf16* pb0 = &Blds[(lane & 15) * 40 + klaneB];
    const bf16* pb1 = &Blds[((lane & 15) + 16) * 40 + klaneB];
    b0.h[0] = *(const v8bf*)(pb0);      b0.h[1] = *(const v8bf*)(pb0 + 16);
    b1.h[0] = *(const v8bf*)(pb1);      b1.h[1] = *(const v8bf*)(pb1 + 16);

    acc0 = __builtin_amdgcn_wmma_f32_16x16x32_bf16(false, a.v, false, b0.v,
                                                   (short)0, acc0, false, false);
    acc1 = __builtin_amdgcn_wmma_f32_16x16x32_bf16(false, a.v, false, b1.v,
                                                   (short)0, acc1, false, false);
    __syncthreads();
  }

  const int ncol0 = n0 + (lane & 15);
  float bs0 = bias[ncol0];
  float bs1 = bias[ncol0 + 16];
  #pragma unroll
  for (int j = 0; j < 8; ++j) {
    int row = m0 + (lane >> 4) * 8 + j;
    xk[(size_t)row * W_SZ + ncol0]      = acc0[j] + bs0;
    xk[(size_t)row * W_SZ + ncol0 + 16] = acc1[j] + bs1;
  }
}

// ---------------------------------------------------------------------------
// Phase 2: persistent scan. 64 WGs = 16 levels x 4 batch-quarters.
// Weight fragments live in VGPRs (128/wave); A staged via TDM tensor DMA;
// per-quarter (16 WG) device barrier between steps.
// ---------------------------------------------------------------------------
__launch_bounds__(320)
__global__ void onlstm_scan(const float* __restrict__ xk, const bf16* __restrict__ RT,
                            bf16* __restrict__ h0buf, bf16* __restrict__ h1buf,
                            unsigned* __restrict__ bar, float* __restrict__ out) {
  __shared__ bf16  Alds[16 * 520];     // 16 rows x 512 K, stride 520 (1040 B)
  __shared__ float zlds[16 * 160];
  __shared__ float fmv[16], imv[16];
  __shared__ float cst[16 * 32];

  const int wg    = blockIdx.x;
  const int lvl   = wg >> 2;
  const int q     = wg & 3;
  const int r0    = q * 16;
  const int tid   = threadIdx.x;
  const int wid   = tid >> 5;
  const int lane  = tid & 31;
  const int nloc  = lane & 15;
  const int klaneA = (lane >> 4) * 8;
  const int klaneB = (lane >> 4) * 8;

  int colbase;
  switch (wid) {
    case 0: colbase = 0;               break;   // f-master
    case 1: colbase = 16;              break;   // i-master
    case 2: colbase = 32   + lvl * 32; break;   // f_gate lo
    case 3: colbase = 48   + lvl * 32; break;   // f_gate hi
    case 4: colbase = 544  + lvl * 32; break;   // i_gate lo
    case 5: colbase = 560  + lvl * 32; break;   // i_gate hi
    case 6: colbase = 1056 + lvl * 32; break;   // o_gate lo
    case 7: colbase = 1072 + lvl * 32; break;   // o_gate hi
    case 8: colbase = 1568 + lvl * 32; break;   // c_in lo
    default: colbase = 1584 + lvl * 32; break;  // c_in hi
  }
  const int col = colbase + nloc;
  const bf16* rcol = RT + (size_t)col * 512;

  // Preload ALL weight fragments for this wave into registers (16 x 8 VGPRs).
  BF16x16 bfrag[16];
  #pragma unroll
  for (int kk = 0; kk < 16; ++kk) {
    const bf16* pb = rcol + kk * 32 + klaneB;   // K = klaneB+[0,8) and klaneB+16+[0,8)
    bfrag[kk].h[0] = *(const v8bf*)(pb);
    bfrag[kk].h[1] = *(const v8bf*)(pb + 16);
  }

  for (int i = tid; i < 512; i += 320) cst[i] = 0.0f;

  // --- TDM descriptors for the per-step A stage (16 x 512 bf16 tile) ------
  // group1: data_size=2B(code1), pad_enable, pad_interval=7 (256 DW = 1 row),
  //         pad_amount=3 (4 DW = 16 B) -> LDS row stride 1040 B.
  const v8u g1 = { 0x07D10000u,
                   (unsigned)(512u << 16),            // tensor_dim0 = 512
                   (unsigned)(16u  << 16),            // tensor_dim1 = 16
                   (unsigned)(512u << 16),            // tile_dim0 = 512
                   16u,                               // tile_dim1 = 16
                   512u,                              // tensor_dim0_stride
                   0u, 0u };
  unsigned lds_a = (unsigned)(size_t)(void*)&Alds[0];
  unsigned long long ga0 = (unsigned long long)(size_t)(h0buf + (size_t)r0 * 512);
  unsigned long long ga1 = (unsigned long long)(size_t)(h1buf + (size_t)r0 * 512);
  const v4u g0_even = { 1u, lds_a, (unsigned)ga0,
                        (unsigned)((ga0 >> 32) & 0x01FFFFFFu) | 0x80000000u };
  const v4u g0_odd  = { 1u, lds_a, (unsigned)ga1,
                        (unsigned)((ga1 >> 32) & 0x01FFFFFFu) | 0x80000000u };

  for (int t = 0; t < T_SZ; ++t) {
    bf16* hnxt = (t & 1) ? h0buf : h1buf;

    // hoist xk loads: independent of h, overlap with DMA + WMMA
    float xkv[8];
    #pragma unroll
    for (int j = 0; j < 8; ++j) {
      int brow = r0 + (lane >> 4) * 8 + j;
      xkv[j] = xk[((size_t)brow * T_SZ + t) * W_SZ + col];
    }

    // stage A via Tensor Data Mover (one DMA per workgroup, wave 0 issues)
    if (wid == 0) {
      v4u g0 = (t & 1) ? g0_odd : g0_even;
      asm volatile("tensor_load_to_lds %0, %1" :: "s"(g0), "s"(g1) : "memory");
      __builtin_amdgcn_s_wait_tensorcnt(0);
    }
    __syncthreads();

    // z tile = h @ R : 16 WMMAs, two accumulator chains
    v8f accA = {}; v8f accB = {};
    #pragma unroll
    for (int kk = 0; kk < 16; kk += 2) {
      BF16x16 a0, a1;
      const bf16* pa0 = &Alds[nloc * 520 + kk * 32 + klaneA];
      a0.h[0] = *(const v8bf*)(pa0);       a0.h[1] = *(const v8bf*)(pa0 + 16);
      const bf16* pa1 = pa0 + 32;
      a1.h[0] = *(const v8bf*)(pa1);       a1.h[1] = *(const v8bf*)(pa1 + 16);
      accA = __builtin_amdgcn_wmma_f32_16x16x32_bf16(false, a0.v, false, bfrag[kk].v,
                                                     (short)0, accA, false, false);
      accB = __builtin_amdgcn_wmma_f32_16x16x32_bf16(false, a1.v, false, bfrag[kk + 1].v,
                                                     (short)0, accB, false, false);
    }

    #pragma unroll
    for (int j = 0; j < 8; ++j) {
      int m = (lane >> 4) * 8 + j;
      zlds[m * 160 + wid * 16 + nloc] = accA[j] + accB[j] + xkv[j];
    }
    __syncthreads();

    // master gates: cumulative softmax over 16 levels, value at `lvl`
    if (tid < 32) {
      int typ = tid >> 4;
      int r   = tid & 15;
      float v[16];
      float mx = -3.4e38f;
      #pragma unroll
      for (int j = 0; j < 16; ++j) { v[j] = zlds[r * 160 + typ * 16 + j]; mx = fmaxf(mx, v[j]); }
      float s = 0.f;
      #pragma unroll
      for (int j = 0; j < 16; ++j) { v[j] = __expf(v[j] - mx); s += v[j]; }
      float inv = 1.0f / s;
      float a2 = 0.f;
      if (typ == 0) {
        for (int j = 0; j <= lvl; ++j) a2 += v[j];
        fmv[r] = a2 * inv;
      } else {
        for (int j = lvl; j < 16; ++j) a2 += v[j];
        imv[r] = a2 * inv;
      }
    }
    __syncthreads();

    // element-wise ON-LSTM update for (16 rows x 32 units)
    for (int idx = tid; idx < 512; idx += 320) {
      int r = idx >> 5, u = idx & 31;
      float zf = zlds[r * 160 +  32 + u];
      float zi = zlds[r * 160 +  64 + u];
      float zo = zlds[r * 160 +  96 + u];
      float zc = zlds[r * 160 + 128 + u];
      float fg = 1.f / (1.f + __expf(-zf));
      float ig = 1.f / (1.f + __expf(-zi));
      float og = 1.f / (1.f + __expf(-zo));
      float ci = 1.f - 2.f / (__expf(2.f * zc) + 1.f);
      float fm = fmv[r], im = imv[r];
      float ov = fm * im;
      float cl = cst[idx];
      float co = ov * (fg * cl + ig * ci) + (fm - ov) * cl + (im - ov) * ci;
      float ho = og * (1.f - 2.f / (__expf(2.f * co) + 1.f));
      cst[idx] = co;
      int brow = r0 + r;
      int gu   = lvl * 32 + u;
      out[((size_t)brow * T_SZ + t) * U_SZ + gu] = ho;
      hnxt[brow * U_SZ + gu] = (bf16)ho;
    }

    // per-quarter device barrier (16 WGs share h rows r0..r0+15)
    __threadfence();
    __syncthreads();
    if (tid == 0) {
      unsigned* c = &bar[q * T_SZ + t];
      __hip_atomic_fetch_add(c, 1u, __ATOMIC_RELEASE, __HIP_MEMORY_SCOPE_AGENT);
      while (__hip_atomic_load(c, __ATOMIC_ACQUIRE, __HIP_MEMORY_SCOPE_AGENT) < GRP) {
        __builtin_amdgcn_s_sleep(1);
      }
    }
    __syncthreads();
    __threadfence();
  }
}

// ---------------------------------------------------------------------------
extern "C" void kernel_launch(void* const* d_in, const int* in_sizes, int n_in,
                              void* d_out, int out_size, void* d_ws, size_t ws_size,
                              hipStream_t stream) {
  const float* x    = (const float*)d_in[0];
  const float* Wk   = (const float*)d_in[1];
  const float* R    = (const float*)d_in[2];
  const float* bias = (const float*)d_in[3];

  char* ws = (char*)d_ws;
  unsigned* bar = (unsigned*)(ws + OFF_BAR);
  bf16* h0b = (bf16*)(ws + OFF_H0);
  bf16* h1b = (bf16*)(ws + OFF_H1);
  bf16* RT  = (bf16*)(ws + OFF_RT);
  float* xk = (float*)(ws + OFF_XK);

  hipMemsetAsync(bar, 0, 4 * T_SZ * sizeof(unsigned), stream);
  hipMemsetAsync(h0b, 0, (size_t)B_SZ * U_SZ * sizeof(bf16), stream);

  onlstm_init_rt<<<(W_SZ * 512) / 256, 256, 0, stream>>>(R, RT);
  onlstm_xk_gemm<<<512 * 65, 256, 0, stream>>>(x, Wk, bias, xk);
  onlstm_scan<<<NWG, 320, 0, stream>>>(xk, RT, h0b, h1b, bar, (float*)d_out);
}
